// Model_8297876815951
// MI455X (gfx1250) — matrix operand order repaired and checked by reading, not compiled, not run on hardware
//
#include <hip/hip_runtime.h>
#include <cmath>
#include <stdint.h>

typedef __attribute__((ext_vector_type(16))) _Float16 v16h;
typedef __attribute__((ext_vector_type(8)))  _Float16 v8h;
typedef __attribute__((ext_vector_type(8)))  float    v8f;
typedef __attribute__((ext_vector_type(4)))  int      v4i;
typedef _Float16 f16;

#define B_    8
#define L_    512
#define NV_   32
#define PRED_ 96
#define D_    128
#define PL_   16
#define E_    8
#define H_    512
#define P_    64
#define N_    16384
#define KDP_  8192   // D*P

__device__ inline v16h concat8(v8h lo, v8h hi) {
  return __builtin_shufflevector(lo, hi, 0,1,2,3,4,5,6,7,8,9,10,11,12,13,14,15);
}

// ---- CDNA5 async global->LDS staging (ASYNCcnt path), with safe fallback ----
// Probe result (round 2): param 1 is 'int __vector(4) *' (generic), so pass v4i*.
#if __has_builtin(__builtin_amdgcn_global_load_async_to_lds_b128)
#define ASYNC_CP16(l, g) \
  __builtin_amdgcn_global_load_async_to_lds_b128((v4i*)(g), (v4i*)(l), 0, 0)
#if __has_builtin(__builtin_amdgcn_s_wait_asynccnt)
#define ASYNC_WAIT() __builtin_amdgcn_s_wait_asynccnt(0)
#else
#define ASYNC_WAIT() asm volatile("s_wait_asynccnt 0x0" ::: "memory")
#endif
#else
#define ASYNC_CP16(l, g) (*(v8h*)(l) = *(const v8h*)(g))
#define ASYNC_WAIT() ((void)0)
#endif

// ---------------- K0: per-(b,v) mean / stdev ----------------
__global__ void stats_kernel(const float* __restrict__ x,
                             float* __restrict__ mean, float* __restrict__ stdev) {
  int bv = blockIdx.x; int b = bv / NV_, v = bv % NV_;
  __shared__ float s1[256], s2[256];
  float a = 0.f, q = 0.f;
  for (int l = threadIdx.x; l < L_; l += 256) {
    float t = x[(b * L_ + l) * NV_ + v];
    a += t; q += t * t;
  }
  s1[threadIdx.x] = a; s2[threadIdx.x] = q; __syncthreads();
  for (int s = 128; s > 0; s >>= 1) {
    if ((int)threadIdx.x < s) { s1[threadIdx.x] += s1[threadIdx.x + s]; s2[threadIdx.x] += s2[threadIdx.x + s]; }
    __syncthreads();
  }
  if (threadIdx.x == 0) {
    float m = s1[0] / (float)L_;
    float var = s2[0] / (float)L_ - m * m;
    mean[bv] = m;
    stdev[bv] = sqrtf(var + 1e-5f);
  }
}

// ---------------- K1: patch embed + sinusoid PE ----------------
__global__ void embed_kernel(const float* __restrict__ x, const float* __restrict__ mean,
                             const float* __restrict__ stdev, const float* __restrict__ Wp,
                             float* __restrict__ xf, f16* __restrict__ xh) {
  int n = blockIdx.x;                   // n = (b*NV+v)*P + p
  int bv = n / P_; int p = n % P_;
  int b = bv / NV_, v = bv % NV_;
  __shared__ float tk[PL_];
  int tid = threadIdx.x;
  if (tid < PL_) {
    int idx = p * 8 + tid; if (idx > L_ - 1) idx = L_ - 1;   // edge pad
    tk[tid] = (x[(b * L_ + idx) * NV_ + v] - mean[bv]) / stdev[bv];
  }
  __syncthreads();
  int d = tid;                          // 128 threads
  float acc = 0.f;
#pragma unroll
  for (int j = 0; j < PL_; j++) acc += tk[j] * Wp[j * D_ + d];
  int i = d >> 1;
  float div = expf(-(float)(2 * i) * (9.210340371976184f / (float)D_)); // ln(10000)
  float ang = (float)p * div;
  acc += (d & 1) ? cosf(ang) : sinf(ang);
  xf[(size_t)n * D_ + d] = acc;
  xh[(size_t)n * D_ + d] = (f16)acc;
}

// ---------------- K2: weight transposes to f16 ----------------
__global__ void w1t_kernel(const float* __restrict__ W1, f16* __restrict__ W1T) { // (E,D,H)->(E,H,D)
  int idx = blockIdx.x * 256 + threadIdx.x;
  if (idx >= E_ * D_ * H_) return;
  int e = idx / (H_ * D_); int r = idx % (H_ * D_); int h = r / D_; int d = r % D_;
  W1T[idx] = (f16)W1[((size_t)e * D_ + d) * H_ + h];
}
__global__ void w2t_kernel(const float* __restrict__ W2, f16* __restrict__ W2T) { // (E,H,D)->(E,D,H)
  int idx = blockIdx.x * 256 + threadIdx.x;
  if (idx >= E_ * D_ * H_) return;
  int e = idx / (D_ * H_); int r = idx % (D_ * H_); int d = r / H_; int h = r % H_;
  W2T[idx] = (f16)W2[((size_t)e * H_ + h) * D_ + d];
}
__global__ void wht_kernel(const float* __restrict__ Wh, f16* __restrict__ WhT) { // (8192,96)->(96,8192)
  int idx = blockIdx.x * 256 + threadIdx.x;
  if (idx >= PRED_ * KDP_) return;
  int o = idx / KDP_; int k = idx % KDP_;
  WhT[idx] = (f16)Wh[(size_t)k * PRED_ + o];
}

// ---------------- K3: gating (softmax, top-2, aux accum) ----------------
__global__ void gate_kernel(const float* __restrict__ xf, const float* __restrict__ Wg,
                            float* __restrict__ cw, float* __restrict__ gacc) {
  __shared__ float sb[2 * E_ + 1];
  int tid = threadIdx.x;
  if (tid < 2 * E_ + 1) sb[tid] = 0.f;
  __syncthreads();
  int n = blockIdx.x * 256 + tid;
  if (n < N_) {
    float lg[E_];
#pragma unroll
    for (int e = 0; e < E_; e++) lg[e] = 0.f;
    for (int d = 0; d < D_; d++) {
      float xv = xf[(size_t)n * D_ + d];
#pragma unroll
      for (int e = 0; e < E_; e++) lg[e] += xv * Wg[d * E_ + e];
    }
    float mx = lg[0];
#pragma unroll
    for (int e = 1; e < E_; e++) mx = fmaxf(mx, lg[e]);
    float pr[E_]; float se = 0.f;
#pragma unroll
    for (int e = 0; e < E_; e++) { pr[e] = expf(lg[e] - mx); se += pr[e]; }
    float inv = 1.f / se;
    float lse = mx + logf(se);
    int i1 = -1, i2 = -1; float m1 = -1e30f, m2 = -1e30f;
#pragma unroll
    for (int e = 0; e < E_; e++) {
      float p = pr[e] * inv; pr[e] = p;
      if (p > m1) { m2 = m1; i2 = i1; m1 = p; i1 = e; }
      else if (p > m2) { m2 = p; i2 = e; }
    }
#pragma unroll
    for (int e = 0; e < E_; e++) {
      float c = (e == i1) ? m1 : ((e == i2) ? m2 : 0.f);
      cw[(size_t)n * E_ + e] = c;
      atomicAdd(&sb[e], pr[e]);
      atomicAdd(&sb[E_ + e], (e == i1 || e == i2) ? 1.f : 0.f);
    }
    atomicAdd(&sb[2 * E_], lse * lse);
  }
  __syncthreads();
  if (tid < 2 * E_ + 1) atomicAdd(&gacc[tid], sb[tid]);
}

// ---------------- K4: dense MoE, WMMA f16, async double-buffered staging ----------------
#define CH_     64
#define SW1_RS  136   // 128 K + pad (halves)
#define SW2_RS  72    // 64 K + pad
#define SH_RS   72
#define SW1_BYTES (CH_ * SW1_RS * 2)      // 17408 per buffer
#define SW2_BYTES (D_ * SW2_RS * 2)       // 18432 per buffer
#define SH_OFF    (2 * SW1_BYTES + 2 * SW2_BYTES)        // 71680
#define SCW_OFF   (SH_OFF + 128 * SH_RS * 2)             // 90112
#define SMEM_BYTES (SCW_OFF + 128 * E_ * 4)              // 94208

__device__ inline void stage_chunk(const f16* __restrict__ W1T, const f16* __restrict__ W2T,
                                   f16* dW1, f16* dW2, int e, int ch, int tid) {
  for (int i = tid; i < CH_ * 16; i += 256) {            // 64 rows x 16 segs of 8 halves
    int r = i >> 4, sg = i & 15;
    const f16* g = &W1T[((size_t)e * H_ + ch * CH_ + r) * D_ + sg * 8];
    f16* l = &dW1[r * SW1_RS + sg * 8];
    ASYNC_CP16(l, g);
  }
  for (int i = tid; i < D_ * 8; i += 256) {              // 128 rows x 8 segs
    int r = i >> 3, sg = i & 7;
    const f16* g = &W2T[((size_t)e * D_ + r) * H_ + ch * CH_ + sg * 8];
    f16* l = &dW2[r * SW2_RS + sg * 8];
    ASYNC_CP16(l, g);
  }
}

__launch_bounds__(256, 1)
__global__ void moe_kernel(const f16* __restrict__ xh, const f16* __restrict__ W1T,
                           const f16* __restrict__ W2T, const float* __restrict__ b1,
                           const float* __restrict__ b2, const float* __restrict__ cw,
                           f16* __restrict__ yflat) {
  extern __shared__ __attribute__((aligned(32))) char smem[];
  f16*   sW1 = (f16*)smem;                       // [2][CH_][SW1_RS]
  f16*   sW2 = (f16*)(smem + 2 * SW1_BYTES);     // [2][D_][SW2_RS]
  f16*   sH  = (f16*)(smem + SH_OFF);            // [128][SH_RS]
  float* sCW = (float*)(smem + SCW_OFF);         // [128][E_]

  int tid  = threadIdx.x;
  int lane = tid & 31;
  int w    = tid >> 5;             // 8 waves
  int m0   = blockIdx.x * 128;
  int ln16 = lane & 15;
  int hi16 = lane >> 4;            // 0 or 1
  int koff = hi16 * 8;

  for (int i = tid; i < 128 * E_; i += 256)
    sCW[i] = cw[(size_t)m0 * E_ + i];

  // persistent A fragments: this wave's 16 token rows, K = 0..127
  int arow = w * 16 + ln16;
  v16h xa[4];
#pragma unroll
  for (int ks = 0; ks < 4; ks++) {
    const f16* pb = xh + (size_t)(m0 + arow) * D_ + ks * 32 + koff;
    xa[ks] = concat8(*(const v8h*)pb, *(const v8h*)(pb + 16));
  }

  v8f Y[8];
#pragma unroll
  for (int t = 0; t < 8; t++)
#pragma unroll
    for (int j = 0; j < 8; j++) Y[t][j] = 0.f;

  // prologue: stage step 0 into buffer 0
  stage_chunk(W1T, W2T, sW1, sW2, 0, 0, tid);

#pragma unroll 1
  for (int s = 0; s < E_ * (H_ / CH_); s++) {
    int buf = s & 1;
    f16* cW1 = sW1 + buf * (CH_ * SW1_RS);
    f16* cW2 = sW2 + buf * (D_ * SW2_RS);

    ASYNC_WAIT();            // our async loads into `buf` are complete
    __syncthreads();         // all waves' loads complete; prev readers of buf^1 done

    if (s + 1 < E_ * (H_ / CH_)) {
      int ns = s + 1;
      f16* nW1 = sW1 + (ns & 1) * (CH_ * SW1_RS);
      f16* nW2 = sW2 + (ns & 1) * (D_ * SW2_RS);
      stage_chunk(W1T, W2T, nW1, nW2, ns >> 3, ns & 7, tid);   // overlap with compute
    }

    int e = s >> 3, ch = s & 7;

    // GEMM1: H_chunk(16 rows/wave x 64) = X(16x128) @ W1chunk(128x64)
#pragma unroll
    for (int t = 0; t < 4; t++) {
      v8f acc;
#pragma unroll
      for (int j = 0; j < 8; j++) acc[j] = 0.f;
#pragma unroll
      for (int ks = 0; ks < 4; ks++) {
        const f16* pb = &cW1[(t * 16 + ln16) * SW1_RS + ks * 32 + hi16 * 8];
        v16h bf = concat8(*(const v8h*)pb, *(const v8h*)(pb + 16));
        acc = __builtin_amdgcn_wmma_f32_16x16x32_f16(false, xa[ks], false, bf,
                                                     (short)0, acc, false, false);
      }
      int ncol = t * 16 + ln16;
      float bias = b1[e * H_ + ch * CH_ + ncol];
#pragma unroll
      for (int j = 0; j < 8; j++) {
        int m = w * 16 + hi16 * 8 + j;
        float xv = acc[j] + bias;
        float g = 0.5f * xv * (1.0f + erff(xv * 0.70710678118f));   // exact gelu
        sH[m * SH_RS + ncol] = (f16)(g * sCW[m * E_ + e]);  // fold gate weight into rows
      }
    }

    // GEMM2: Y(16 rows/wave x 128) += Hc(16x64) @ W2chunk(64x128)
#pragma unroll
    for (int ks = 0; ks < 2; ks++) {
      const f16* pa = &sH[(w * 16 + ln16) * SH_RS + ks * 32 + koff];
      v16h af = concat8(*(const v8h*)pa, *(const v8h*)(pa + 16));
#pragma unroll
      for (int t = 0; t < 8; t++) {
        const f16* pb = &cW2[(t * 16 + ln16) * SW2_RS + ks * 32 + hi16 * 8];
        v16h bf = concat8(*(const v8h*)pb, *(const v8h*)(pb + 16));
        Y[t] = __builtin_amdgcn_wmma_f32_16x16x32_f16(false, af, false, bf,
                                                      (short)0, Y[t], false, false);
      }
    }
  }

  // epilogue: + sum_e cw*b2, store in head-flat f16 layout yflat[r][d*64+p]
#pragma unroll
  for (int t = 0; t < 8; t++) {
    int ncol = t * 16 + ln16;
#pragma unroll
    for (int j = 0; j < 8; j++) {
      int m = w * 16 + hi16 * 8 + j;
      float val = Y[t][j];
#pragma unroll
      for (int e = 0; e < E_; e++) val += sCW[m * E_ + e] * b2[e * D_ + ncol];
      int n = m0 + m;
      int r = n >> 6, p = n & 63;
      yflat[(size_t)r * KDP_ + ncol * 64 + p] = (f16)val;
    }
  }
}

// ---------------- K5: head GEMM + denorm + transpose ----------------
__global__ void head_kernel(const f16* __restrict__ yflat, const f16* __restrict__ WhT,
                            const float* __restrict__ bh, const float* __restrict__ mean,
                            const float* __restrict__ stdev, float* __restrict__ out) {
  int lane = threadIdx.x & 31;
  int ln16 = lane & 15, hi16 = lane >> 4;
  int m0 = blockIdx.x * 16;     // token-row (r) tile
  int n0 = blockIdx.y * 16;     // output (o) tile
  v8f acc;
#pragma unroll
  for (int j = 0; j < 8; j++) acc[j] = 0.f;
  const f16* arow = yflat + (size_t)(m0 + ln16) * KDP_;
  const f16* brow = WhT   + (size_t)(n0 + ln16) * KDP_;
  for (int k0 = 0; k0 < KDP_; k0 += 32) {
    const f16* pa = arow + k0 + hi16 * 8;
    v16h af = concat8(*(const v8h*)pa, *(const v8h*)(pa + 16));
    const f16* pb = brow + k0 + hi16 * 8;
    v16h bf = concat8(*(const v8h*)pb, *(const v8h*)(pb + 16));
    acc = __builtin_amdgcn_wmma_f32_16x16x32_f16(false, af, false, bf,
                                                 (short)0, acc, false, false);
  }
  int o = n0 + ln16;
#pragma unroll
  for (int j = 0; j < 8; j++) {
    int r = m0 + hi16 * 8 + j;      // r = b*NV + v
    int b = r >> 5, v = r & 31;
    float val = (acc[j] + bh[o]) * stdev[r] + mean[r];
    out[((size_t)b * PRED_ + o) * NV_ + v] = val;
  }
}

// ---------------- K6: aux loss ----------------
__global__ void aux_kernel(const float* __restrict__ gacc, float* __restrict__ out) {
  if (threadIdx.x == 0 && blockIdx.x == 0) {
    float bal = 0.f;
    for (int e = 0; e < E_; e++) bal += (gacc[e] / (float)N_) * (gacc[E_ + e] / (float)N_);
    bal *= (float)E_ / 2.0f;                 // E * sum / TOPN
    float z = gacc[2 * E_] / (float)N_;
    out[B_ * PRED_ * NV_] = 0.01f * bal + 0.001f * z;
  }
}

// ---------------- launch ----------------
extern "C" void kernel_launch(void* const* d_in, const int* in_sizes, int n_in,
                              void* d_out, int out_size, void* d_ws, size_t ws_size,
                              hipStream_t stream) {
  const float* x_enc  = (const float*)d_in[0];
  const float* W_patch= (const float*)d_in[4];
  const float* Wg     = (const float*)d_in[5];
  const float* W1     = (const float*)d_in[6];
  const float* b1     = (const float*)d_in[7];
  const float* W2     = (const float*)d_in[8];
  const float* b2     = (const float*)d_in[9];
  const float* W_head = (const float*)d_in[10];
  const float* b_head = (const float*)d_in[11];
  float* out = (float*)d_out;

  char* ws = (char*)d_ws;
  float* mean  = (float*)(ws + 0);                 // 256 f32
  float* stdev = (float*)(ws + 1024);              // 256 f32
  float* gacc  = (float*)(ws + 2048);              // 17 f32
  float* cw    = (float*)(ws + 4096);              // 16384*8 f32   = 512 KB
  float* xf    = (float*)(ws + 528384);            // 16384*128 f32 = 8 MB
  f16*   xh    = (f16*)  (ws + 8916992);           // 16384*128 f16 = 4 MB
  f16*   W1T   = (f16*)  (ws + 13111296);          // 8*512*128 f16 = 1 MB
  f16*   W2T   = (f16*)  (ws + 14159872);          // 8*128*512 f16 = 1 MB
  f16*   WhT   = (f16*)  (ws + 15208448);          // 96*8192 f16   = 1.5 MB
  f16*   yflat = (f16*)  (ws + 16781312);          // 256*8192 f16  = 4 MB

  (void)hipMemsetAsync(gacc, 0, 32 * sizeof(float), stream);

  stats_kernel<<<B_ * NV_, 256, 0, stream>>>(x_enc, mean, stdev);
  embed_kernel<<<N_, 128, 0, stream>>>(x_enc, mean, stdev, W_patch, xf, xh);
  w1t_kernel<<<(E_ * D_ * H_) / 256, 256, 0, stream>>>(W1, W1T);
  w2t_kernel<<<(E_ * D_ * H_) / 256, 256, 0, stream>>>(W2, W2T);
  wht_kernel<<<(PRED_ * KDP_) / 256, 256, 0, stream>>>(W_head, WhT);
  gate_kernel<<<N_ / 256, 256, 0, stream>>>(xf, Wg, cw, gacc);
  moe_kernel<<<N_ / 128, 256, SMEM_BYTES, stream>>>(xh, W1T, W2T, b1, b2, cw, yflat);
  head_kernel<<<dim3((B_ * NV_) / 16, PRED_ / 16), 32, 0, stream>>>(yflat, WhT, b_head, mean, stdev, out);
  aux_kernel<<<1, 1, 0, stream>>>(gacc, out);
}
